// ABC_2D_Agnostic_67095979098604
// MI455X (gfx1250) — compile-verified, matrix operand order checked by reading, not run on hardware
//
#include <hip/hip_runtime.h>

typedef __attribute__((ext_vector_type(16))) _Float16 v16h;
typedef __attribute__((ext_vector_type(8)))  float    v8f;

#define BATCH   32
#define CIN     32
#define PIX     4096        // 64*64
#define KS      9
#define KOUT    64
#define RDIM    (CIN * KS)  // 288 = GEMM inner dimension
#define BLOCK_N 128         // pixels (columns) per workgroup
#define NFRAG   (4 * 9)     // 4 row-tiles x 9 k-steps of A fragments

// K-index mapping for the CDNA5 16-bit A/B matrix VGPR layout (ISA 7.12.2):
// VGPR v<4 covers K 0..7, v>=4 covers K 16..23; hi lane group adds +8.
__device__ __forceinline__ constexpr int kmap_base(int h) {
    int v = h >> 1, rb = h & 1;
    return (v < 4 ? 2 * v : 16 + 2 * (v - 4)) + rb;
}

__global__ __launch_bounds__(256)
void abc2d_gather_wmma(const float* __restrict__ x,
                       const float* __restrict__ w,
                       const int*   __restrict__ gidx,
                       float*       __restrict__ out)
{
    __shared__ v16h wfrag[NFRAG * 32];     // A fragments, f16, WMMA lane layout (36 KB)

    const int tid    = threadIdx.x;
    const int lane   = tid & 31;           // wave32
    const int wave   = tid >> 5;           // 8 waves
    const int laneHi = lane >> 4;

    const int wg = blockIdx.x;
    const int b  = wg >> 5;                // 32 WGs per batch (4096 / 128)
    const int p0 = (wg & 31) * BLOCK_N;

    // ---- stage weights into WMMA A-fragment layout, fp32 -> fp16 ----
    // Inner-dim permutation: WMMA slot t = ks*32 + K  <->  reference r = K*9 + ks
    // (i.e. channel c = K, neighbor slot s = ks).  288 = 9*32, exact split.
    {
        _Float16* wh = (_Float16*)wfrag;
        for (int f = tid; f < NFRAG * 32 * 16; f += 256) {   // 18432 = 256*72, uniform
            int h    = f & 15;
            int ln   = (f >> 4) & 31;
            int frag = f >> 9;
            int mt   = frag / 9;
            int ks   = frag - mt * 9;
            int c    = kmap_base(h) + ((ln >> 4) << 3);      // channel = WMMA K slot
            int M    = ln & 15;                              // output-channel within tile
            wh[f] = (_Float16)w[(mt * 16 + M) * RDIM + c * KS + ks];
        }
    }
    __syncthreads();

    const int pl = wave * 16 + (lane & 15);   // pixel within WG tile (= B-matrix N)

    // ---- load this lane's 9 neighbor ids once (36 contiguous bytes) ----
    int q[KS];
    {
        const int* ip = gidx + (size_t)(p0 + pl) * KS;
        #pragma unroll
        for (int s = 0; s < KS; ++s) q[s] = ip[s];
    }

    // Base pointer already folds in batch and the hi-lane-group +8 channels;
    // per-gather channel offset becomes a compile-time immediate.
    const float* xg = x + (size_t)b * CIN * PIX + (size_t)(laneHi << 3) * PIX;

    v8f acc0 = {}, acc1 = {}, acc2 = {}, acc3 = {};

    #pragma unroll
    for (int ks = 0; ks < KS; ++ks) {
        // ---- gather B fragment: 16 imm-offset loads off one base ----
        const float* gp = xg + q[ks];
        v16h bf;
        #pragma unroll
        for (int h = 0; h < 16; ++h)
            bf[h] = (_Float16)gp[kmap_base(h) * PIX];        // global_load_b32, imm offset

        // ---- A fragments from LDS (32 contiguous bytes per lane) ----
        v16h a0 = wfrag[(0 * 9 + ks) * 32 + lane];
        v16h a1 = wfrag[(1 * 9 + ks) * 32 + lane];
        v16h a2 = wfrag[(2 * 9 + ks) * 32 + lane];
        v16h a3 = wfrag[(3 * 9 + ks) * 32 + lane];

        acc0 = __builtin_amdgcn_wmma_f32_16x16x32_f16(false, a0, false, bf, (short)0, acc0, false, false);
        acc1 = __builtin_amdgcn_wmma_f32_16x16x32_f16(false, a1, false, bf, (short)0, acc1, false, false);
        acc2 = __builtin_amdgcn_wmma_f32_16x16x32_f16(false, a2, false, bf, (short)0, acc2, false, false);
        acc3 = __builtin_amdgcn_wmma_f32_16x16x32_f16(false, a3, false, bf, (short)0, acc3, false, false);
    }

    // ---- store D: out[b, k, p], C/D layout lane0-15 M=j, lane16-31 M=j+8 ----
    float* outb = out + (size_t)b * KOUT * PIX;
    const int pcol = p0 + pl;
    #pragma unroll
    for (int j = 0; j < 8; ++j) {
        int row = j + (laneHi << 3);
        outb[(0 * 16 + row) * PIX + pcol] = acc0[j];
        outb[(1 * 16 + row) * PIX + pcol] = acc1[j];
        outb[(2 * 16 + row) * PIX + pcol] = acc2[j];
        outb[(3 * 16 + row) * PIX + pcol] = acc3[j];
    }
}

extern "C" void kernel_launch(void* const* d_in, const int* in_sizes, int n_in,
                              void* d_out, int out_size, void* d_ws, size_t ws_size,
                              hipStream_t stream) {
    const float* x   = (const float*)d_in[0];   // (B, C, H, W) fp32
    const float* w   = (const float*)d_in[1];   // (K, C*KS)    fp32
    const int*   idx = (const int*)d_in[2];     // (1, H, W, KS) int32
    float*       out = (float*)d_out;           // (B, K, H, W) fp32

    dim3 grid((BATCH * PIX) / BLOCK_N);         // 1024 workgroups
    dim3 block(256);                            // 8 wave32s
    hipLaunchKernelGGL(abc2d_gather_wmma, grid, block, 0, stream, x, w, idx, out);
    (void)in_sizes; (void)n_in; (void)out_size; (void)d_ws; (void)ws_size;
}
